// MHA_65429531787938
// MI455X (gfx1250) — matrix operand order repaired and checked by reading, not compiled, not run on hardware
//
#include <hip/hip_runtime.h>
#include <hip/hip_bf16.h>
#include <stdint.h>

// ---------------------------------------------------------------------------
// MI455X (gfx1250) fused MHA with batch-axis softmax quirk.
// Matmuls via v_wmma_f32_16x16x32_bf16; attention K/V tiles staged in LDS
// with GLOBAL_LOAD_ASYNC_TO_LDS (ASYNCcnt) and consumed via ds_load.
// ---------------------------------------------------------------------------

typedef __attribute__((ext_vector_type(16))) __bf16 v16bf;
typedef __attribute__((ext_vector_type(8)))  __bf16 v8bf;
typedef __attribute__((ext_vector_type(8)))  float  v8f;
typedef int gvec4i __attribute__((vector_size(16)));   // matches builtin param

typedef __attribute__((address_space(1))) gvec4i* glob_v4i_ptr;
typedef __attribute__((address_space(3))) gvec4i* lds_v4i_ptr;

#define DEV __device__ __forceinline__

constexpr int Bc = 2, Sq = 2048, Dm = 1024, Hn = 16, HDim = 64;
constexpr int Mrows = Bc * Sq;                 // 4096
constexpr float INV_SCALE = 1.0f / 32.0f;      // 1/sqrt(D_Q) = 1/32

// Build a 16-element bf16 fragment from two 16-byte loads.
DEV v16bf ld16(const __bf16* p0, const __bf16* p1) {
  v8bf lo = *reinterpret_cast<const v8bf*>(p0);
  v8bf hi = *reinterpret_cast<const v8bf*>(p1);
  return __builtin_shufflevector(lo, hi, 0, 1, 2, 3, 4, 5, 6, 7,
                                 8, 9, 10, 11, 12, 13, 14, 15);
}

// A-matrix fragment (16x32, row per lane): lane<16 -> K {c..c+7, c+16..c+23},
// c = kbase (lanes 0-15) or kbase+8 (lanes 16-31).  [ISA 7.12.2]
DEV v16bf load_frag_a(const __bf16* rowp, int kbase, int lane) {
  int c = kbase + ((lane < 16) ? 0 : 8);
  return ld16(rowp + c, rowp + c + 16);
}

// B-matrix fragment (32x16, column per lane) from a K-contiguous (transposed)
// operand: K {c..c+7, c+16..c+23}, c = kbase (lanes 0-15) / kbase+8 (16-31).
DEV v16bf load_frag_b(const __bf16* rowp, int kbase, int lane) {
  int c = kbase + ((lane < 16) ? 0 : 8);
  return ld16(rowp + c, rowp + c + 16);
}

DEV v8f wmma_bf16(v16bf a, v16bf b, v8f c) {
  return __builtin_amdgcn_wmma_f32_16x16x32_bf16(
      /*neg_a=*/false, a, /*neg_b=*/false, b,
      /*c_mod=*/(short)0, c, /*reuse_a=*/false, /*reuse_b=*/false);
}

// ---- gfx1250 async global->LDS copy (ASYNCcnt), with safe fallback --------
DEV void stage_async(const __bf16* gsrc, __bf16* ldst) {
#if __has_builtin(__builtin_amdgcn_global_load_async_to_lds_b128)
  __builtin_amdgcn_global_load_async_to_lds_b128(
      (glob_v4i_ptr)(uintptr_t)gsrc,
      (lds_v4i_ptr)(uint32_t)(uintptr_t)ldst,
      0, 0);
#else
  *reinterpret_cast<v8bf*>(ldst) = *reinterpret_cast<const v8bf*>(gsrc);
#endif
}

DEV void wait_async() {
#if __has_builtin(__builtin_amdgcn_s_wait_asynccnt)
  __builtin_amdgcn_s_wait_asynccnt(0);
#else
  asm volatile("s_wait_asynccnt 0x0" ::: "memory");
#endif
}

// ---------------------------------------------------------------------------
// Elementwise cast f32 -> bf16
// ---------------------------------------------------------------------------
__global__ void cast_to_bf16(const float* __restrict__ in,
                             __bf16* __restrict__ out, int n) {
  int i = blockIdx.x * blockDim.x + threadIdx.x;
  if (i < n) out[i] = (__bf16)in[i];
}

// ---------------------------------------------------------------------------
// Weight transpose + cast: w[k][n] (f32, 1024x1024) -> wt[n][k] (bf16)
// ---------------------------------------------------------------------------
__global__ void transpose_cast(const float* __restrict__ w,
                               __bf16* __restrict__ wt) {
  int i = blockIdx.x * blockDim.x + threadIdx.x;  // Dm*Dm elements
  int k = i >> 10;
  int n = i & (Dm - 1);
  wt[(size_t)n * Dm + k] = (__bf16)w[i];
}

// ---------------------------------------------------------------------------
// GEMM  C[M,N] = A[M,K] x B[K,N], A bf16 row-major, Bt[N,K] bf16 transposed.
// M=4096, N=K=1024.  8 waves -> 128x128 block; each wave owns a 32x64 tile
// (8 accumulators -> 8 WMMAs per 12 b128 loads per k-step).
// MODE 0: f32 row-major out; MODE 1: Q/K head layout; MODE 2: V^T layout.
// ---------------------------------------------------------------------------
enum { OUT_F32 = 0, OUT_QK = 1, OUT_VT = 2 };

template <int MODE>
__global__ __launch_bounds__(256) void gemm_bf16(const __bf16* __restrict__ A,
                                                 const __bf16* __restrict__ Bt,
                                                 void* __restrict__ outp) {
  constexpr int Kd = Dm, Nd = Dm;
  const int lane = threadIdx.x & 31;
  const int wave = threadIdx.x >> 5;
  const int wm = wave >> 1, wn = wave & 1;
  const int m0 = blockIdx.x * 128 + wm * 32;
  const int n0 = blockIdx.y * 128 + wn * 64;
  const int l15 = lane & 15;

  const __bf16* arow0 = A + (size_t)(m0 + l15) * Kd;
  const __bf16* arow1 = A + (size_t)(m0 + 16 + l15) * Kd;
  const __bf16* brow[4];
  for (int t = 0; t < 4; ++t) brow[t] = Bt + (size_t)(n0 + t * 16 + l15) * Kd;

  v8f acc[2][4] = {};
  for (int k = 0; k < Kd; k += 32) {
    if (k + 64 < Kd) {  // software prefetch -> global_prefetch_b8
      __builtin_prefetch(arow0 + k + 64, 0, 0);
      __builtin_prefetch(brow[0] + k + 64, 0, 0);
    }
    v16bf a0 = load_frag_a(arow0, k, lane);
    v16bf a1 = load_frag_a(arow1, k, lane);
    for (int t = 0; t < 4; ++t) {
      v16bf b = load_frag_b(brow[t], k, lane);
      acc[0][t] = wmma_bf16(a0, b, acc[0][t]);
      acc[1][t] = wmma_bf16(a1, b, acc[1][t]);
    }
  }

  // C/D layout: VGPR r, lanes 0-15 -> (M=r, N=lane); lanes 16-31 -> (M=r+8).
  const int rhi = (lane >> 4) << 3;
  for (int mi = 0; mi < 2; ++mi)
    for (int ni = 0; ni < 4; ++ni)
      for (int r = 0; r < 8; ++r) {
        int m = m0 + mi * 16 + r + rhi;
        int n = n0 + ni * 16 + l15;
        float v = acc[mi][ni][r];
        if constexpr (MODE == OUT_F32) {
          ((float*)outp)[(size_t)m * Nd + n] = v;
        } else {
          int b = m >> 11, s = m & (Sq - 1);
          int h = n >> 6, hd = n & (HDim - 1);
          __bf16 bv = (__bf16)v;
          if constexpr (MODE == OUT_QK)
            ((__bf16*)outp)[((size_t)(b * Hn + h) * Sq + s) * HDim + hd] = bv;
          else
            ((__bf16*)outp)[((size_t)(b * Hn + h) * HDim + hd) * Sq + s] = bv;
        }
      }
}

// ---------------------------------------------------------------------------
// Fused attention: per (head, 64-query tile) workgroup, 4 waves x 16 queries.
// K/V tiles (shared by all 4 waves) are double-buffered in LDS via async
// global->LDS copies; fragments come from ds_load_b128.  Streams 32-key
// tiles: both batches' score tiles -> exact 2-element batch-axis softmax
// (max-shifted) -> LDS repack to A-fragments -> P@V accumulate.
// ---------------------------------------------------------------------------
__global__ __launch_bounds__(128) void attn_fused(const __bf16* __restrict__ Qm,
                                                  const __bf16* __restrict__ Km,
                                                  const __bf16* __restrict__ Vt,
                                                  __bf16* __restrict__ vals) {
  // K tile: [buf][b][key 0..31][hd 0..63]; V tile: [buf][b][hd 0..63][key 0..31]
  __shared__ __align__(16) __bf16 Kt[2][2 * 32 * 64];
  __shared__ __align__(16) __bf16 Vl[2][2 * 64 * 32];
  __shared__ __align__(16) __bf16 scratch[4][2][16][40];  // 80B rows (16B mult)

  const int tid = threadIdx.x;
  const int lane = tid & 31;
  const int wv = tid >> 5;
  const int h = blockIdx.y;
  const int q0 = blockIdx.x * 64 + wv * 16;
  const int l15 = lane & 15;
  const int rhi = (lane >> 4) << 3;

  // ---- async staging of one 32-key K/V tile pair (16 KB) -----------------
  auto stage_tiles = [&](int kt, int buf) {
    for (int i = 0; i < 4; ++i) {
      int e = (tid + i * 128) * 8;      // flat element offset, 8 bf16 chunks
      int b = e >> 11;                  // / (32*64)
      int rem = e & 2047;
      {
        int row = rem >> 6, hd = rem & 63;
        const __bf16* g = Km + ((size_t)(b * Hn + h) * Sq + kt + row) * HDim + hd;
        stage_async(g, &Kt[buf][e]);
      }
      {
        int hd = rem >> 5, kc = rem & 31;
        const __bf16* g = Vt + ((size_t)(b * Hn + h) * HDim + hd) * Sq + kt + kc;
        stage_async(g, &Vl[buf][e]);
      }
    }
  };

  // Q fragments, held in registers for the whole key sweep: [batch][kchunk]
  v16bf qf[2][2];
  for (int b = 0; b < 2; ++b) {
    const __bf16* qrow = Qm + ((size_t)(b * Hn + h) * Sq + q0 + l15) * HDim;
    qf[b][0] = load_frag_a(qrow, 0, lane);
    qf[b][1] = load_frag_a(qrow, 32, lane);
  }

  v8f acc[2][4] = {};  // [batch][hd tile of 16]

  stage_tiles(0, 0);
  wait_async();
  __syncthreads();

  for (int kt = 0; kt < Sq; kt += 32) {
    const int cur = (kt >> 5) & 1;
    if (kt + 32 < Sq) stage_tiles(kt + 32, cur ^ 1);  // prefetch next tile

    // ---- scores: S_b = Q_b K_b^T (16q x 32k), both batches, from LDS -----
    v8f sc[2][2];
    for (int b = 0; b < 2; ++b)
      for (int j = 0; j < 2; ++j) {
        const __bf16* krow = &Kt[cur][(b * 32 + j * 16 + l15) * 64];
        v16bf k0 = load_frag_b(krow, 0, lane);
        v16bf k1 = load_frag_b(krow, 32, lane);
        v8f s = {};
        s = wmma_bf16(qf[b][0], k0, s);
        s = wmma_bf16(qf[b][1], k1, s);
        sc[b][j] = s;
      }

    // ---- softmax over BATCH axis (2 elements), elementwise ---------------
    for (int j = 0; j < 2; ++j)
      for (int r = 0; r < 8; ++r) {
        float s0 = sc[0][j][r], s1 = sc[1][j][r];
        float mx = fmaxf(s0, s1);
        float e0 = __expf((s0 - mx) * INV_SCALE);
        float e1 = __expf((s1 - mx) * INV_SCALE);
        float rd = 1.0f / (e0 + e1);
        int row = r + rhi;
        int col = j * 16 + l15;
        scratch[wv][0][row][col] = (__bf16)(e0 * rd);
        scratch[wv][1][row][col] = (__bf16)(e1 * rd);
      }
    __builtin_amdgcn_wave_barrier();  // LDS is wave-in-order; pin scheduling

    // ---- repack probabilities as A-fragments (16q x 32k) -----------------
    v16bf af[2];
    af[0] = load_frag_a(&scratch[wv][0][l15][0], 0, lane);
    af[1] = load_frag_a(&scratch[wv][1][l15][0], 0, lane);
    __builtin_amdgcn_wave_barrier();

    // ---- P @ V: V^T rows (hd-major, keys contiguous) from LDS ------------
    for (int b = 0; b < 2; ++b)
      for (int t = 0; t < 4; ++t) {
        const __bf16* vrow = &Vl[cur][(b * 64 + t * 16 + l15) * 32];
        v16bf vf = load_frag_b(vrow, 0, lane);
        acc[b][t] = wmma_bf16(af[b], vf, acc[b][t]);
      }

    wait_async();      // this wave's prefetch for buf^1 has landed
    __syncthreads();   // all waves done reading cur, next buffer ready
  }

  // ---- write vals in [b, s, h*64+hd] bf16 layout for the out projection --
  for (int b = 0; b < 2; ++b)
    for (int t = 0; t < 4; ++t)
      for (int r = 0; r < 8; ++r) {
        int row = q0 + r + rhi;
        int col = h * HDim + t * 16 + l15;
        vals[((size_t)b * Sq + row) * Dm + col] = (__bf16)acc[b][t][r];
      }
}

// ---------------------------------------------------------------------------
extern "C" void kernel_launch(void* const* d_in, const int* in_sizes, int n_in,
                              void* d_out, int out_size, void* d_ws,
                              size_t ws_size, hipStream_t stream) {
  (void)in_sizes; (void)n_in; (void)out_size; (void)ws_size;
  const float* x  = (const float*)d_in[0];
  const float* wq = (const float*)d_in[1];
  const float* wk = (const float*)d_in[2];
  const float* wv = (const float*)d_in[3];
  const float* wo = (const float*)d_in[4];

  // Workspace carve (~48 MiB total, 256B-aligned chunks).
  char* ws = (char*)d_ws;
  size_t off = 0;
  auto carve = [&](size_t bytes) {
    void* p = ws + off;
    off += (bytes + 255) & ~(size_t)255;
    return p;
  };
  __bf16* xb   = (__bf16*)carve((size_t)Mrows * Dm * 2);
  __bf16* wqt  = (__bf16*)carve((size_t)Dm * Dm * 2);
  __bf16* wkt  = (__bf16*)carve((size_t)Dm * Dm * 2);
  __bf16* wvt  = (__bf16*)carve((size_t)Dm * Dm * 2);
  __bf16* wot  = (__bf16*)carve((size_t)Dm * Dm * 2);
  __bf16* Qm   = (__bf16*)carve((size_t)Mrows * Dm * 2);
  __bf16* Km   = (__bf16*)carve((size_t)Mrows * Dm * 2);
  __bf16* Vtm  = (__bf16*)carve((size_t)Mrows * Dm * 2);
  __bf16* valn = (__bf16*)carve((size_t)Mrows * Dm * 2);

  // 1) cast activations, transpose+cast weights (all L2-resident afterwards)
  int nx = Mrows * Dm;
  cast_to_bf16<<<nx / 256, 256, 0, stream>>>(x, xb, nx);
  int nw = Dm * Dm;
  transpose_cast<<<nw / 256, 256, 0, stream>>>(wq, wqt);
  transpose_cast<<<nw / 256, 256, 0, stream>>>(wk, wkt);
  transpose_cast<<<nw / 256, 256, 0, stream>>>(wv, wvt);
  transpose_cast<<<nw / 256, 256, 0, stream>>>(wo, wot);

  // 2) Q/K/V projections (4096x1024x1024 each) straight into head layouts
  dim3 ggrid(Mrows / 128, Dm / 128);
  gemm_bf16<OUT_QK><<<ggrid, 256, 0, stream>>>(xb, wqt, Qm);
  gemm_bf16<OUT_QK><<<ggrid, 256, 0, stream>>>(xb, wkt, Km);
  gemm_bf16<OUT_VT><<<ggrid, 256, 0, stream>>>(xb, wvt, Vtm);

  // 3) fused attention with batch-axis softmax (no score materialization)
  dim3 agrid(Sq / 64, Hn);
  attn_fused<<<agrid, 128, 0, stream>>>(Qm, Km, Vtm, valn);

  // 4) output projection -> f32 d_out
  gemm_bf16<OUT_F32><<<ggrid, 256, 0, stream>>>(valn, wot, d_out);
}
